// MultiheadAttention_74122545594860
// MI455X (gfx1250) — matrix operand order repaired and checked by reading, not compiled, not run on hardware
//
#include <hip/hip_runtime.h>
#include <hip/hip_bf16.h>

// ---------------------------------------------------------------------------
// Multi-head causal attention forward for MI455X (gfx1250, wave32, WMMA).
// fp32 -> bf16 operands, V_WMMA_F32_16X16X32_BF16, fp32 accumulation.
// ~412 GFLOP vs ~0.5 GB HBM traffic -> matrix-unit bound.
// CDNA5 specifics used: v_wmma_f32_16x16x32_bf16, global_load_async_to_lds_b128
// (ASYNCcnt) for non-transposing tile stages, global_prefetch, wave32 shuffles.
// ---------------------------------------------------------------------------

#define NUM_HEADS 16
#define DIM_HEAD  128
#define D_MODEL   2048
#define BATCH     4
#define SEQ       2048
#define ROWS      (BATCH * SEQ)   // 8192

#define USE_ASYNC_LDS 1

typedef __attribute__((ext_vector_type(16))) __bf16 bf16x16;
typedef __attribute__((ext_vector_type(8)))  float  f32x8;

union FragAB {
    bf16x16        v;
    uint4          q[2];   // two 16B chunks
    unsigned short us[16];
};

__device__ __forceinline__ unsigned short f2bf(float f) {
    union { float f; unsigned u; } x; x.f = f;
    unsigned r = x.u + 0x7fffu + ((x.u >> 16) & 1u);   // round-to-nearest-even
    return (unsigned short)(r >> 16);
}
__device__ __forceinline__ float bf2f(unsigned short h) {
    union { unsigned u; float f; } x; x.u = ((unsigned)h) << 16;
    return x.f;
}
__device__ __forceinline__ f32x8 zero8() {
    f32x8 z;
#pragma unroll
    for (int i = 0; i < 8; ++i) z[i] = 0.0f;
    return z;
}
__device__ __forceinline__ float rmax16(float v) {
#pragma unroll
    for (int m = 1; m < 16; m <<= 1) v = fmaxf(v, __shfl_xor(v, m, 32));
    return v;
}
__device__ __forceinline__ float rsum16(float v) {
#pragma unroll
    for (int m = 1; m < 16; m <<= 1) v += __shfl_xor(v, m, 32);
    return v;
}

// Per-lane async 32B row-chunk copy global -> LDS (two b128 ops, ASYNCcnt).
// VDST gives the LDS address and VADDR the global one, each plus the offset;
// 16 bytes per lane per op. Generic LDS pointer's low 32 bits are the LDS
// byte offset (flat aperture mapping), which is what VDST wants.
__device__ __forceinline__ void async_copy32(const unsigned short* gsrc,
                                             unsigned short* ldst) {
#if USE_ASYNC_LDS
    unsigned           d = (unsigned)(unsigned long long)ldst;
    unsigned long long s = (unsigned long long)gsrc;
    asm volatile("global_load_async_to_lds_b128 %0, %1, off"
                 :: "v"(d), "v"(s) : "memory");
    asm volatile("global_load_async_to_lds_b128 %0, %1, off offset:16"
                 :: "v"(d), "v"(s) : "memory");
#else
    ((uint4*)ldst)[0] = ((const uint4*)gsrc)[0];
    ((uint4*)ldst)[1] = ((const uint4*)gsrc)[1];
#endif
}
__device__ __forceinline__ void async_wait() {
#if USE_ASYNC_LDS
    asm volatile("s_wait_asynccnt 0x0" ::: "memory");
#endif
}

// ---------------------------------------------------------------------------
// fp32 -> bf16 conversion
// ---------------------------------------------------------------------------
__global__ void cvt_bf16(const float* __restrict__ src,
                         unsigned short* __restrict__ dst, int n) {
    int i = blockIdx.x * blockDim.x + threadIdx.x;
    if (i < n) dst[i] = f2bf(src[i]);
}

// ---------------------------------------------------------------------------
// bf16 GEMM: Out[M=8192, N] = A[8192, 2048] @ B[2048, N] + bias
// 128x128 block tile, 8 waves as 4(M) x 2(N); wave = 32x64 (2x4 tiles),
// K-step 32. A tile staged via async-to-LDS (row-major copy); B staged with a
// manual transpose to [n][k]. 16-bit WMMA operand layouts:
//   A: element e of lane l holds K = (e&7) + (e>>3)*16 + (l>>4)*8
//   B: element e of lane l holds K = (e&7) + (e>>3)*16 + (l>>4)*8, N = l&15
// ---------------------------------------------------------------------------
template <bool BF16OUT>
__global__ __launch_bounds__(256) void gemm_bf16(
    const unsigned short* __restrict__ A,
    const unsigned short* __restrict__ B,
    const float* __restrict__ bias,
    void* __restrict__ Out, int N) {
    __shared__ __align__(16) unsigned short Alds[128 * 40];
    __shared__ __align__(16) unsigned short Blds[128 * 40];   // [n][k]

    const int tid  = threadIdx.x;
    const int lane = tid & 31;
    const int wid  = tid >> 5;
    const int wm   = (wid >> 1) * 32;       // wave M offset inside tile
    const int wn   = (wid & 1) * 64;        // wave N offset inside tile
    const int mbase = blockIdx.y * 128;
    const int nbase = blockIdx.x * 128;

    f32x8 acc[2][4];
#pragma unroll
    for (int i = 0; i < 2; ++i)
#pragma unroll
        for (int j = 0; j < 4; ++j) acc[i][j] = zero8();

    const int arow = tid >> 1;              // 0..127
    const int acol = (tid & 1) * 16;        // 0 or 16
    const int brow = tid >> 3;              // k 0..31
    const int bcol = (tid & 7) * 16;        // n sub-offset

    for (int kb = 0; kb < 2048; kb += 32) {
        __syncthreads();
        // stage A tile (128x32) via CDNA5 async global->LDS copy
        async_copy32(A + (size_t)(mbase + arow) * 2048 + kb + acol,
                     Alds + arow * 40 + acol);
        { // stage B tile (32x128) transposed into [n][k]
            const unsigned short* src =
                B + (size_t)(kb + brow) * N + nbase + bcol;
            FragAB t;
            t.q[0] = ((const uint4*)src)[0];
            t.q[1] = ((const uint4*)src)[1];
#pragma unroll
            for (int i = 0; i < 16; ++i)
                Blds[(bcol + i) * 40 + brow] = t.us[i];
        }
        if (kb + 32 < 2048) { // CDNA5 global_prefetch of next tiles
            __builtin_prefetch(A + (size_t)(mbase + arow) * 2048 + kb + 32 + acol, 0, 0);
            __builtin_prefetch(B + (size_t)(kb + 32 + brow) * N + nbase + bcol, 0, 0);
        }
        async_wait();
        __syncthreads();

        FragAB af[2];
#pragma unroll
        for (int mt = 0; mt < 2; ++mt) {
            int off = (wm + mt * 16 + (lane & 15)) * 40 + ((lane >> 4) * 8);
            af[mt].q[0] = *(const uint4*)(Alds + off);
            af[mt].q[1] = *(const uint4*)(Alds + off + 16);
        }
        FragAB bf[4];
#pragma unroll
        for (int nt = 0; nt < 4; ++nt) {
            int off = (wn + nt * 16 + (lane & 15)) * 40 + ((lane >> 4) * 8);
            bf[nt].q[0] = *(const uint4*)(Blds + off);
            bf[nt].q[1] = *(const uint4*)(Blds + off + 16);
        }
#pragma unroll
        for (int mt = 0; mt < 2; ++mt)
#pragma unroll
            for (int nt = 0; nt < 4; ++nt)
                acc[mt][nt] = __builtin_amdgcn_wmma_f32_16x16x32_bf16(
                    false, af[mt].v, false, bf[nt].v, (short)0, acc[mt][nt],
                    false, false);
    }

    // epilogue: C/D layout -> row M = v + 8*(lane>>4), col N = lane&15
#pragma unroll
    for (int mt = 0; mt < 2; ++mt)
#pragma unroll
        for (int nt = 0; nt < 4; ++nt) {
            int   n  = nbase + wn + nt * 16 + (lane & 15);
            float bv = bias[n];
            int   m0 = mbase + wm + mt * 16 + (lane >> 4) * 8;
#pragma unroll
            for (int v = 0; v < 8; ++v) {
                float  val = acc[mt][nt][v] + bv;
                size_t idx = (size_t)(m0 + v) * N + n;
                if constexpr (BF16OUT)
                    ((unsigned short*)Out)[idx] = f2bf(val);
                else
                    ((float*)Out)[idx] = val;
            }
        }
}

// ---------------------------------------------------------------------------
// RoPE on Q (reference ropes Q only), reshape to [B,H,S,Dh] bf16
// ---------------------------------------------------------------------------
__global__ void rope_q(const unsigned short* __restrict__ Qtmp,
                       unsigned short* __restrict__ Qh) {
    int idx = blockIdx.x * blockDim.x + threadIdx.x;  // B*S*H*64 = 2^23
    if (idx >= BATCH * SEQ * NUM_HEADS * 64) return;
    int i = idx & 63;
    int h = (idx >> 6) & 15;
    int s = (idx >> 10) & 2047;
    int b = idx >> 21;
    size_t src = ((size_t)(b * SEQ + s)) * D_MODEL + h * DIM_HEAD + 2 * i;
    float xe = bf2f(Qtmp[src]);
    float xo = bf2f(Qtmp[src + 1]);
    // inv_freq = 10000^(-2i/128) = exp(-ln(10000)/64 * i)
    float ang = (float)s * __expf(-0.14391156831f * (float)i);
    float sn, cs;
    __sincosf(ang, &sn, &cs);
    size_t dst = (((size_t)(b * NUM_HEADS + h)) * SEQ + s) * DIM_HEAD + 2 * i;
    Qh[dst]     = f2bf(xe * cs - xo * sn);
    Qh[dst + 1] = f2bf(xe * sn + xo * cs);
}

__global__ void split_kv(const unsigned short* __restrict__ KV,
                         unsigned short* __restrict__ Kh,
                         unsigned short* __restrict__ Vh) {
    int idx = blockIdx.x * blockDim.x + threadIdx.x;  // ROWS*D_MODEL = 2^24
    if (idx >= ROWS * D_MODEL) return;
    int d = idx & 127;
    int h = (idx >> 7) & 15;
    int s = (idx >> 11) & 2047;
    int b = idx >> 22;
    size_t row = (size_t)(b * SEQ + s);
    size_t dst = (((size_t)(b * NUM_HEADS + h)) * SEQ + s) * DIM_HEAD + d;
    Kh[dst] = KV[row * 4096 + h * 128 + d];
    Vh[dst] = KV[row * 4096 + 2048 + h * 128 + d];
}

// ---------------------------------------------------------------------------
// Flash attention, causal. Block = (q-tile of 128) x head x batch, 8 waves.
// Wave owns 16 query rows; Q fragments (16x128) live in registers.
// Per 32-key tile:
//   - K tile (32x128) staged row-major to LDS via async copy (shared by all
//     8 waves: kills the 8x redundant global reads of the previous version)
//   - V tile staged transposed [d][key] with manual stores
//   - 8 WMMAs Q@K^T, online softmax (16-lane shuffle reductions),
//     P transposed through per-wave LDS, 8 WMMAs P@V.
// ---------------------------------------------------------------------------
__global__ __launch_bounds__(256) void flash_attn(
    const unsigned short* __restrict__ Qh,
    const unsigned short* __restrict__ Kh,
    const unsigned short* __restrict__ Vh,
    unsigned short* __restrict__ Outc) {   // [ROWS, D_MODEL] bf16
    __shared__ __align__(16) unsigned short Klds[32 * 136];    // [key][d]
    __shared__ __align__(16) unsigned short Vt[128 * 40];      // [d][key]
    __shared__ __align__(16) unsigned short Plds[8 * 16 * 40]; // per-wave P

    const int lane  = threadIdx.x & 31;
    const int wid   = threadIdx.x >> 5;
    const int qtile = blockIdx.x * 128;
    const int h     = blockIdx.y;
    const int b     = blockIdx.z;
    const size_t headbase =
        ((size_t)(b * NUM_HEADS + h)) * (size_t)SEQ * DIM_HEAD;

    const int qw = qtile + wid * 16;       // this wave's first query row

    FragAB qf[4];                          // Q 16x128 as 4 K-chunks of 32
    {
        const unsigned short* qrow =
            Qh + headbase + (size_t)(qw + (lane & 15)) * DIM_HEAD;
#pragma unroll
        for (int kc = 0; kc < 4; ++kc) {
            int d0 = kc * 32 + (lane >> 4) * 8;
            qf[kc].q[0] = ((const uint4*)(qrow + d0))[0];
            qf[kc].q[1] = ((const uint4*)(qrow + d0 + 16))[0];
        }
    }

    f32x8 Oacc[8];
#pragma unroll
    for (int i = 0; i < 8; ++i) Oacc[i] = zero8();
    float mrun[8], lrun[8];
#pragma unroll
    for (int v = 0; v < 8; ++v) { mrun[v] = -1e30f; lrun[v] = 0.0f; }
    const float scale = 0.08838834764831845f;  // 1/sqrt(128)

    const int tkey = threadIdx.x >> 3;         // staging: key row 0..31
    const int tdp  = (threadIdx.x & 7) * 16;   // staging: d sub-offset

    const int kend = qtile + 128;              // causal bound (block-uniform)
    for (int kb = 0; kb < kend; kb += 32) {
        __syncthreads();
        // K tile: async row-major copy global -> LDS (ASYNCcnt path)
        async_copy32(Kh + headbase + (size_t)(kb + tkey) * DIM_HEAD + tdp,
                     Klds + tkey * 136 + tdp);
        { // V tile transposed into Vt[d][key] (manual: async can't transpose)
            const unsigned short* src =
                Vh + headbase + (size_t)(kb + tkey) * DIM_HEAD + tdp;
            FragAB t;
            t.q[0] = ((const uint4*)src)[0];
            t.q[1] = ((const uint4*)src)[1];
#pragma unroll
            for (int i = 0; i < 16; ++i) Vt[(tdp + i) * 40 + tkey] = t.us[i];
        }
        async_wait();
        __syncthreads();

        // scores S = Q @ K^T, two 16-key subtiles; K frags from LDS
        f32x8 sacc[2];
#pragma unroll
        for (int sub = 0; sub < 2; ++sub) {
            sacc[sub] = zero8();
            const unsigned short* krow =
                Klds + (sub * 16 + (lane & 15)) * 136;
#pragma unroll
            for (int kc = 0; kc < 4; ++kc) {
                FragAB bm;
                int d0 = kc * 32 + (lane >> 4) * 8;
                bm.q[0] = *(const uint4*)(krow + d0);
                bm.q[1] = *(const uint4*)(krow + d0 + 16);
                sacc[sub] = __builtin_amdgcn_wmma_f32_16x16x32_bf16(
                    false, qf[kc].v, false, bm.v, (short)0, sacc[sub], false,
                    false);
            }
        }

        // causal mask + online softmax (row M = v + 8*(lane>>4), col = lane&15)
        float p[2][8];
#pragma unroll
        for (int v = 0; v < 8; ++v) {
            int   q  = qw + v + (lane >> 4) * 8;
            float s0 = sacc[0][v] * scale;
            float s1 = sacc[1][v] * scale;
            int   k0 = kb + (lane & 15);
            if (k0 > q)      s0 = -1e30f;
            if (k0 + 16 > q) s1 = -1e30f;
            float tm    = rmax16(fmaxf(s0, s1));
            float mn    = fmaxf(mrun[v], tm);
            float alpha = __expf(mrun[v] - mn);
            float p0    = __expf(s0 - mn);
            float p1    = __expf(s1 - mn);
            p[0][v] = p0;
            p[1][v] = p1;
            lrun[v] = lrun[v] * alpha + rsum16(p0 + p1);
            mrun[v] = mn;
#pragma unroll
            for (int ds_ = 0; ds_ < 8; ++ds_) Oacc[ds_][v] *= alpha;
        }

        // transpose P (D-layout) -> A-layout via per-wave LDS
        unsigned short* pw = Plds + wid * 16 * 40;
#pragma unroll
        for (int v = 0; v < 8; ++v) {
            int M = v + (lane >> 4) * 8;
            pw[M * 40 + (lane & 15)]      = f2bf(p[0][v]);
            pw[M * 40 + 16 + (lane & 15)] = f2bf(p[1][v]);
        }
        FragAB af;
        {
            int off = (lane & 15) * 40 + (lane >> 4) * 8;
            af.q[0] = *(const uint4*)(pw + off);
            af.q[1] = *(const uint4*)(pw + off + 16);
        }

        // O += P @ V  (8 d-subtiles)
#pragma unroll
        for (int dsub = 0; dsub < 8; ++dsub) {
            FragAB bm;
            int off = (dsub * 16 + (lane & 15)) * 40 + (lane >> 4) * 8;
            bm.q[0] = *(const uint4*)(Vt + off);
            bm.q[1] = *(const uint4*)(Vt + off + 16);
            Oacc[dsub] = __builtin_amdgcn_wmma_f32_16x16x32_bf16(
                false, af.v, false, bm.v, (short)0, Oacc[dsub], false, false);
        }
    }

    // normalize and store to [B*S, H*Dh] bf16 for the Wo GEMM
#pragma unroll
    for (int v = 0; v < 8; ++v) {
        float  rl   = 1.0f / lrun[v];
        int    srow = qw + v + (lane >> 4) * 8;
        size_t orow = ((size_t)(b * SEQ + srow)) * D_MODEL + h * DIM_HEAD;
#pragma unroll
        for (int dsub = 0; dsub < 8; ++dsub)
            Outc[orow + dsub * 16 + (lane & 15)] = f2bf(Oacc[dsub][v] * rl);
    }
}

// ---------------------------------------------------------------------------
// Host-side orchestration (all on `stream`, graph-capture safe)
// ---------------------------------------------------------------------------
extern "C" void kernel_launch(void* const* d_in, const int* in_sizes, int n_in,
                              void* d_out, int out_size, void* d_ws,
                              size_t ws_size, hipStream_t stream) {
    (void)in_sizes; (void)n_in; (void)out_size; (void)ws_size;
    const float* x   = (const float*)d_in[0];
    const float* Wq  = (const float*)d_in[1];
    const float* bq  = (const float*)d_in[2];
    const float* Wkv = (const float*)d_in[3];
    const float* bkv = (const float*)d_in[4];
    const float* Wo  = (const float*)d_in[5];
    const float* bo  = (const float*)d_in[6];

    char* ws = (char*)d_ws;
    unsigned short* xb    = (unsigned short*)ws; ws += (size_t)ROWS * D_MODEL * 2;
    unsigned short* wqb   = (unsigned short*)ws; ws += (size_t)D_MODEL * D_MODEL * 2;
    unsigned short* wkvb  = (unsigned short*)ws; ws += (size_t)D_MODEL * 2 * D_MODEL * 2;
    unsigned short* wob   = (unsigned short*)ws; ws += (size_t)D_MODEL * D_MODEL * 2;
    unsigned short* qtmp  = (unsigned short*)ws; ws += (size_t)ROWS * D_MODEL * 2;
    unsigned short* kvtmp = (unsigned short*)ws; ws += (size_t)ROWS * 2 * D_MODEL * 2;
    unsigned short* qh    = (unsigned short*)ws; ws += (size_t)ROWS * D_MODEL * 2;
    unsigned short* kh    = (unsigned short*)ws; ws += (size_t)ROWS * D_MODEL * 2;
    unsigned short* vh    = (unsigned short*)ws; ws += (size_t)ROWS * D_MODEL * 2;
    unsigned short* attnb = qtmp;  // qtmp dead after rope_q -> reuse (256MB total)

    const int nx = ROWS * D_MODEL;          // 16,777,216
    const int nw = D_MODEL * D_MODEL;       // 4,194,304
    cvt_bf16<<<(nx + 255) / 256, 256, 0, stream>>>(x, xb, nx);
    cvt_bf16<<<(nw + 255) / 256, 256, 0, stream>>>(Wq, wqb, nw);
    cvt_bf16<<<(2 * nw + 255) / 256, 256, 0, stream>>>(Wkv, wkvb, 2 * nw);
    cvt_bf16<<<(nw + 255) / 256, 256, 0, stream>>>(Wo, wob, nw);

    gemm_bf16<true><<<dim3(16, 64), 256, 0, stream>>>(xb, wqb, bq, qtmp, 2048);
    gemm_bf16<true><<<dim3(32, 64), 256, 0, stream>>>(xb, wkvb, bkv, kvtmp, 4096);

    const int nrope = BATCH * SEQ * NUM_HEADS * 64;  // 8,388,608
    rope_q<<<(nrope + 255) / 256, 256, 0, stream>>>(qtmp, qh);
    split_kv<<<(nx + 255) / 256, 256, 0, stream>>>(kvtmp, kh, vh);

    flash_attn<<<dim3(SEQ / 128, NUM_HEADS, BATCH), 256, 0, stream>>>(qh, kh, vh,
                                                                      attnb);

    gemm_bf16<false><<<dim3(16, 64), 256, 0, stream>>>(attnb, wob, bo,
                                                       (void*)d_out, 2048);
}
